// LinearMultiHeadAttention_67224828117015
// MI455X (gfx1250) — matrix operand order repaired and checked by reading, not compiled, not run on hardware
//
#include <hip/hip_runtime.h>

typedef __attribute__((ext_vector_type(16))) __bf16 v16bf;
typedef __attribute__((ext_vector_type(8)))  float  v8f;

#define LDA  40   // bf16 elements per As row: BK(32) + 8 pad -> 80B rows, 16B aligned
#define LDB2 20   // uints per Bs row: 16 K-pairs + 4 pad -> 80B rows, 16B aligned

// ---------------------------------------------------------------------------
// Fragment loaders matching CDNA5 WMMA VGPR layouts (ISA 7.12.2, wave32)
// A 16x32 bf16: lanes 0-15 -> M=lane, K = {0..7,16..23}; lanes 16-31 -> +8
// ---------------------------------------------------------------------------
__device__ __forceinline__ v16bf load_frag_a(const __bf16* As, int rowbase, int lane) {
  const int half = lane >> 4;
  const int m    = lane & 15;
  const __bf16* p = As + (rowbase + m) * LDA + half * 8;
  union { uint4 q[2]; v16bf v; } u;
  u.q[0] = *(const uint4*)(p);        // K = half*8 .. half*8+7
  u.q[1] = *(const uint4*)(p + 16);   // K = 16+half*8 .. +7
  return u.v;
}

// B 32x16 bf16 staged as interleaved K-pairs: Bs[n][p] = {B[2p,n], B[2p+1,n]}
// lanes 0-15 -> N=lane, K={0..7,16..23} (pairs 0..3,8..11); lanes 16-31 -> +8 (pairs 4..7,12..15)
__device__ __forceinline__ v16bf load_frag_b(const unsigned int* Bs, int colbase, int lane) {
  const int half = lane >> 4;
  const int n    = lane & 15;
  const unsigned int* p = Bs + (colbase + n) * LDB2 + half * 4;
  union { uint4 q[2]; v16bf v; } u;
  u.q[0] = *(const uint4*)(p);
  u.q[1] = *(const uint4*)(p + 8);
  return u.v;
}

// ---------------------------------------------------------------------------
// Tiled WMMA GEMM: D[m,n] = sum_k A[m,k] * B[k,n]  (+bias[m], epilogue)
// BM=64, BN=256, BK=32; 256 threads = 8 waves, each wave a 32x64 output tile
// (8 v_wmma per K-step). Register double-buffered global->LDS pipeline +
// global_prefetch two tiles ahead.
// EPI: 0 = bf16 (+bias), 1 = elu(x)+1 -> bf16, 2 = f32 (+bias).
// TRANSA: A given as AT[k][m] row-major (used for context^T in attn apply).
// ---------------------------------------------------------------------------
template <int EPI, bool TRANSA>
__global__ __launch_bounds__(256)
void wmma_gemm(const __bf16* __restrict__ A, long long strideA,
               const __bf16* __restrict__ Bmat, long long strideB,
               const float* __restrict__ bias,
               void* __restrict__ Dv, long long strideD,
               int K, int lda, int ldb, int ldd)
{
  constexpr int BM = 64, BN = 256, BK = 32;
  __shared__ __bf16        As[BM * LDA];     // 5.0 KB
  __shared__ unsigned int  Bs[BN * LDB2];    // 20.0 KB

  const int z = blockIdx.z;
  const __bf16* Ab = A    + (size_t)z * (size_t)strideA;
  const __bf16* Bb = Bmat + (size_t)z * (size_t)strideB;

  const int row0 = blockIdx.y * BM;
  const int col0 = blockIdx.x * BN;
  const int t    = threadIdx.x;
  const int lane = t & 31;
  const int w    = t >> 5;
  const int wm   = (w & 1) * 32;   // 2 waves along M
  const int wn   = (w >> 1) * 64;  // 4 waves along N, 64 cols each

  // staging thread mappings
  const int ar  = t >> 2;          // A row 0..63
  const int acg = (t & 3) * 8;     // A col group
  const int akk = t >> 3;          // TRANSA: k row 0..31
  const int amg = (t & 7) * 8;     // TRANSA: m group 0..56
  const int br2 = t >> 4;          // B K-pair row 0..15
  const int bng = (t & 15) * 16;   // B col group (16 cols)

  uint4 aReg, bLo0, bHi0, bLo1, bHi1;

  auto loadTiles = [&](int k0) {
    if (!TRANSA)
      aReg = *(const uint4*)(Ab + (size_t)(row0 + ar) * lda + k0 + acg);
    else
      aReg = *(const uint4*)(Ab + (size_t)(k0 + akk) * lda + row0 + amg);
    const __bf16* p0 = Bb + (size_t)(k0 + 2 * br2) * ldb + col0 + bng;
    bLo0 = *(const uint4*)(p0);
    bHi0 = *(const uint4*)(p0 + ldb);
    bLo1 = *(const uint4*)(p0 + 8);
    bHi1 = *(const uint4*)(p0 + ldb + 8);
  };
  auto storeTiles = [&]() {
    if (!TRANSA) {
      *(uint4*)(&As[ar * LDA + acg]) = aReg;
    } else {
      const __bf16* pv = (const __bf16*)&aReg;
#pragma unroll
      for (int i = 0; i < 8; ++i) As[(amg + i) * LDA + akk] = pv[i];
    }
    const unsigned short* l0 = (const unsigned short*)&bLo0;
    const unsigned short* h0 = (const unsigned short*)&bHi0;
    const unsigned short* l1 = (const unsigned short*)&bLo1;
    const unsigned short* h1 = (const unsigned short*)&bHi1;
#pragma unroll
    for (int i = 0; i < 8; ++i) {
      Bs[(bng + i)     * LDB2 + br2] = (unsigned int)l0[i] | ((unsigned int)h0[i] << 16);
      Bs[(bng + 8 + i) * LDB2 + br2] = (unsigned int)l1[i] | ((unsigned int)h1[i] << 16);
    }
  };

  v8f acc[2][4] = {};
  const int nt = K / BK;

  loadTiles(0);
  for (int kt = 0; kt < nt; ++kt) {
    storeTiles();
    __syncthreads();
    if (kt + 1 < nt) loadTiles((kt + 1) * BK);   // overlap HBM/L2 with WMMA
    if (!TRANSA && kt + 2 < nt) {                // warm L2 two tiles ahead
      const int kp = (kt + 2) * BK;
      __builtin_prefetch(Ab + (size_t)(row0 + ar) * lda + kp + acg, 0, 1);
      __builtin_prefetch(Bb + (size_t)(kp + 2 * br2) * ldb + col0 + bng, 0, 1);
    }

    v16bf af[2], bfb[4];
#pragma unroll
    for (int i = 0; i < 2; ++i) af[i]  = load_frag_a(As, wm + 16 * i, lane);
#pragma unroll
    for (int j = 0; j < 4; ++j) bfb[j] = load_frag_b(Bs, wn + 16 * j, lane);
#pragma unroll
    for (int i = 0; i < 2; ++i)
#pragma unroll
      for (int j = 0; j < 4; ++j)
        acc[i][j] = __builtin_amdgcn_wmma_f32_16x16x32_bf16(
            false, af[i], false, bfb[j], (short)0, acc[i][j], false, false);
    __syncthreads();
  }

  // ---- epilogue: C/D layout -> lanes 0-15: M=r,N=lane; lanes 16-31: M=r+8 ----
  const int half = lane >> 4;
  const int n    = lane & 15;
#pragma unroll
  for (int i = 0; i < 2; ++i) {
#pragma unroll
    for (int j = 0; j < 4; ++j) {
#pragma unroll
      for (int r = 0; r < 8; ++r) {
        const int gm = row0 + wm + 16 * i + r + 8 * half;
        const int gn = col0 + wn + 16 * j + n;
        float val = acc[i][j][r];
        if (bias) val += bias[gm];
        const size_t off = (size_t)z * (size_t)strideD + (size_t)gm * ldd + gn;
        if (EPI == 2) {
          ((float*)Dv)[off] = val;
        } else if (EPI == 1) {
          const float kp = val > 0.0f ? val + 1.0f : __expf(val);  // elu+1
          ((__bf16*)Dv)[off] = (__bf16)kp;
        } else {
          ((__bf16*)Dv)[off] = (__bf16)val;
        }
      }
    }
  }
}

// ---------------------------------------------------------------------------
// context[bh][d][e] += sum_s Kp[bh,d,s] * V[bh,e,s]   (M=N=64, K=S, split-K)
// B = V^T: K-pairs are contiguous in V rows -> pure uint4 staging.
// ---------------------------------------------------------------------------
__global__ __launch_bounds__(256)
void context_wmma(const __bf16* __restrict__ Kp, const __bf16* __restrict__ V,
                  float* __restrict__ Ctx, int S, int Ssplit)
{
  constexpr int BK = 32;
  __shared__ __bf16       As[64 * LDA];
  __shared__ unsigned int Bs[64 * LDB2];

  const int bh = blockIdx.x;                 // 0..63 (b*H + h)
  const int s_begin = blockIdx.y * Ssplit;
  const __bf16* Ab = Kp + (size_t)bh * 64 * S;
  const __bf16* Vb = V  + (size_t)bh * 64 * S;
  float* Cb = Ctx + (size_t)bh * 4096;

  const int t    = threadIdx.x;
  const int lane = t & 31;
  const int w    = t >> 5;
  const int wm   = (w & 3) * 16;   // 4 waves along M
  const int wn   = (w >> 2) * 32;  // 2 waves along N (2 tiles each)

  const int ar  = t >> 2;          // A row 0..63
  const int acg = (t & 3) * 8;     // A col group
  const int bn  = t >> 2;          // B col (V row) 0..63
  const int bpg = (t & 3) * 4;     // K-pair group

  uint4 aReg, bReg;
  auto loadTiles = [&](int k0) {
    aReg = *(const uint4*)(Ab + (size_t)ar * S + k0 + acg);
    bReg = *(const uint4*)(Vb + (size_t)bn * S + k0 + bpg * 2);
  };

  v8f acc[2] = {};
  const int nt = Ssplit / BK;

  loadTiles(s_begin);
  for (int kt = 0; kt < nt; ++kt) {
    *(uint4*)(&As[ar * LDA + acg])     = aReg;
    *(uint4*)(&Bs[bn * LDB2 + bpg])    = bReg;
    __syncthreads();
    if (kt + 1 < nt) loadTiles(s_begin + (kt + 1) * BK);
    if (kt + 2 < nt) {
      const int kp = s_begin + (kt + 2) * BK;
      __builtin_prefetch(Ab + (size_t)ar * S + kp + acg, 0, 1);
      __builtin_prefetch(Vb + (size_t)bn * S + kp + bpg * 2, 0, 1);
    }

    v16bf af  = load_frag_a(As, wm, lane);
    v16bf bf0 = load_frag_b(Bs, wn, lane);
    v16bf bf1 = load_frag_b(Bs, wn + 16, lane);
    acc[0] = __builtin_amdgcn_wmma_f32_16x16x32_bf16(false, af, false, bf0, (short)0, acc[0], false, false);
    acc[1] = __builtin_amdgcn_wmma_f32_16x16x32_bf16(false, af, false, bf1, (short)0, acc[1], false, false);
    __syncthreads();
  }

  const int half = lane >> 4;
  const int n    = lane & 15;
#pragma unroll
  for (int j = 0; j < 2; ++j)
#pragma unroll
    for (int r = 0; r < 8; ++r) {
      const int gm = wm + r + 8 * half;
      const int gn = wn + 16 * j + n;
      atomicAdd(&Cb[gm * 64 + gn], acc[j][r]);
    }
}

// ---------------------------------------------------------------------------
__global__ void f32_to_bf16(const float* __restrict__ src, __bf16* __restrict__ dst,
                            long long n) {
  long long i = (long long)blockIdx.x * blockDim.x + threadIdx.x;
  const long long stride = (long long)gridDim.x * blockDim.x;
  for (; i < n; i += stride) dst[i] = (__bf16)src[i];
}

__global__ void zero_f32(float* __restrict__ p, long long n) {
  long long i = (long long)blockIdx.x * blockDim.x + threadIdx.x;
  const long long stride = (long long)gridDim.x * blockDim.x;
  for (; i < n; i += stride) p[i] = 0.0f;
}

// ---------------------------------------------------------------------------
extern "C" void kernel_launch(void* const* d_in, const int* in_sizes, int n_in,
                              void* d_out, int out_size, void* d_ws, size_t ws_size,
                              hipStream_t stream) {
  (void)in_sizes; (void)n_in; (void)out_size; (void)ws_size;
  constexpr int B = 4, C = 1024, S = 4096, H = 16, BH = B * H;
  constexpr long long NX = (long long)B * C * S;   // 16.7M
  constexpr long long NW = (long long)C * C;       // 1M

  const float* x  = (const float*)d_in[0];
  const float* wq = (const float*)d_in[1];
  const float* bq = (const float*)d_in[2];
  const float* wk = (const float*)d_in[3];
  const float* bk = (const float*)d_in[4];
  const float* wv = (const float*)d_in[5];
  const float* bv = (const float*)d_in[6];
  const float* wo = (const float*)d_in[7];
  const float* bo = (const float*)d_in[8];
  float* out = (float*)d_out;

  // workspace layout (bf16 intermediates + f32 context accumulator)
  char* ws = (char*)d_ws;
  size_t off = 0;
  auto take = [&](size_t bytes) { char* p = ws + off; off += (bytes + 255) & ~(size_t)255; return p; };
  __bf16* xb    = (__bf16*)take(NX * 2);
  __bf16* qb    = (__bf16*)take(NX * 2);
  __bf16* kpb   = (__bf16*)take(NX * 2);
  __bf16* vb    = (__bf16*)take(NX * 2);
  __bf16* attnb = (__bf16*)take(NX * 2);
  __bf16* wqb   = (__bf16*)take(NW * 2);
  __bf16* wkb   = (__bf16*)take(NW * 2);
  __bf16* wvb   = (__bf16*)take(NW * 2);
  __bf16* wob   = (__bf16*)take(NW * 2);
  float*  ctxf  = (float*)take((size_t)BH * 64 * 64 * 4);
  __bf16* ctxb  = (__bf16*)take((size_t)BH * 64 * 64 * 2);

  // 1) precision conversion: f32 -> bf16
  f32_to_bf16<<<4096, 256, 0, stream>>>(x,  xb,  NX);
  f32_to_bf16<<<1024, 256, 0, stream>>>(wq, wqb, NW);
  f32_to_bf16<<<1024, 256, 0, stream>>>(wk, wkb, NW);
  f32_to_bf16<<<1024, 256, 0, stream>>>(wv, wvb, NW);
  f32_to_bf16<<<1024, 256, 0, stream>>>(wo, wob, NW);
  zero_f32<<<512, 256, 0, stream>>>(ctxf, (long long)BH * 64 * 64);

  // 2) Q/K'/V projections: [1024x1024] x [1024x4096] per batch (bf16 WMMA)
  const dim3 gProj(S / 256, C / 64, B), blk(256);
  const long long sX = (long long)C * S;
  wmma_gemm<0, false><<<gProj, blk, 0, stream>>>(wqb, 0, xb, sX, bq, qb,  sX, C, C, S, S);
  wmma_gemm<1, false><<<gProj, blk, 0, stream>>>(wkb, 0, xb, sX, bk, kpb, sX, C, C, S, S); // elu+1 fused
  wmma_gemm<0, false><<<gProj, blk, 0, stream>>>(wvb, 0, xb, sX, bv, vb,  sX, C, C, S, S);

  // 3) context = K' V^T per (b,h): 64x64xS, split-K x8 with f32 atomics
  context_wmma<<<dim3(BH, 8), blk, 0, stream>>>(kpb, vb, ctxf, S, S / 8);
  f32_to_bf16<<<256, 256, 0, stream>>>(ctxf, ctxb, (long long)BH * 64 * 64);

  // 4) attn^T[e,s] = sum_d ctx[d,e] * q[d,s] per (b,h): A = ctx^T (TRANSA), K=64
  wmma_gemm<0, true><<<dim3(S / 256, 1, BH), blk, 0, stream>>>(
      ctxb, 64 * 64, qb, (long long)64 * S, nullptr, attnb, (long long)64 * S,
      64, 64, S, S);

  // 5) output projection: out = Wo x attn + bo, f32 result
  wmma_gemm<2, false><<<gProj, blk, 0, stream>>>(wob, 0, attnb, sX, bo, out, sX, C, C, S, S);
}
